// LatticeRNN_19516331393521
// MI455X (gfx1250) — matrix operand order repaired and checked by reading, not compiled, not run on hardware
//
#include <hip/hip_runtime.h>
#include <hip/hip_bf16.h>
#include <stddef.h>

// LatticeRNN on MI455X (gfx1250), bf16 WMMA path.
//
// Sizes: GH=GW=9, B=1024, H=256, 3H=768, 4H=1024.
// d_out layout (flat concat, fp32):
//   [0)        out      : 1024
//   [1024)     final_h  : 1024*256
//   [263168)   final_c  : 1024*256
//   [525312)   gh       : 81*1024*256
//   [21758976) gc       : 81*1024*256
//
// If ws_size allows (~101 MB), all weights are pre-converted once per call to
// bf16 in d_ws; the bf16 weight image fits in the 192 MB L2, halving L2-side
// weight traffic (which is replicated across the 64 batch-tile blocks) and
// removing all cvt VALU work from the WMMA inner loops.

#define GHN 9
#define GWN 9
#define BN  1024
#define HN  256
#define K3  768
#define G4  1024

typedef __attribute__((ext_vector_type(16))) __bf16 v16bf;
typedef __attribute__((ext_vector_type(8)))  __bf16 v8bf;
typedef __attribute__((ext_vector_type(8)))  float  v8f;

__device__ __forceinline__ float sigmoidf_(float v) {
  return 1.0f / (1.0f + __expf(-v));
}

// A-fragment (contiguous k order, same as ldB_frag): lane holds row m=lane&15; element e maps
// to K = e + 16*(lane>>4)  -> two contiguous 8-half (16B) runs.
__device__ __forceinline__ v16bf ldsA_frag(const __bf16* row, int off) {
  union { v16bf v; v8bf h[2]; } u;
  u.h[0] = *(const v8bf*)(row + off);
  u.h[1] = *(const v8bf*)(row + off + 8);
  return u.v;
}

// B-fragment: lane holds column n=lane&15; element e maps to K = e + 16*(lane>>4)
// -> 16 contiguous weights per lane.
__device__ __forceinline__ v16bf ldB_frag(const float* wrow) {
  union { v16bf v; __bf16 e[16]; } u;
  const float4* p = (const float4*)wrow;
#pragma unroll
  for (int q4 = 0; q4 < 4; ++q4) {
    float4 f = p[q4];
    u.e[4*q4+0] = (__bf16)f.x;
    u.e[4*q4+1] = (__bf16)f.y;
    u.e[4*q4+2] = (__bf16)f.z;
    u.e[4*q4+3] = (__bf16)f.w;
  }
  return u.v;
}

__device__ __forceinline__ v16bf ldB_frag(const __bf16* wrow) {
  return *(const v16bf*)wrow;   // 32 contiguous bytes -> 2x global_load_b128
}

template <typename WT>
__global__ __launch_bounds__(256)
void lattice_cell_kernel(const float* __restrict__ x,
                         const float* __restrict__ h_ext, const float* __restrict__ c_ext,
                         const float* __restrict__ grid_h, const float* __restrict__ grid_c,
                         const WT*    __restrict__ W_hp, const float* __restrict__ b_hp,
                         const WT*    __restrict__ W_cp, const float* __restrict__ b_cp,
                         const float* __restrict__ W_ih, const WT*    __restrict__ W_hh,
                         const float* __restrict__ b_ih, const float* __restrict__ b_hh,
                         float* __restrict__ gh_out, float* __restrict__ gc_out,
                         int d, int ilo)
{
  __shared__ __bf16 sA [16 * K3];   // comb_h tile, bf16      (24 KB)
  __shared__ __bf16 sAc[16 * K3];   // comb_c tile, bf16      (24 KB)
  __shared__ __bf16 sGA[16 * HN];   // ph result as gates-A   ( 8 KB)

  const int ci   = ilo + (int)blockIdx.y;
  const int cj   = d - ci;
  const int k    = ci * GWN + cj;
  const int row0 = (int)blockIdx.x * 16;     // batch rows [row0, row0+16)
  const int tid  = (int)threadIdx.x;
  const int w    = tid >> 5;                 // wave id 0..7
  const int lane = tid & 31;
  const int ln   = lane & 15;
  const int kh   = lane >> 4;

  const size_t BH = (size_t)BN * HN;

  // Neighbor state sources (nullptr => zeros).
  const float* hL = nullptr; const float* cL = nullptr;
  const float* hU = nullptr; const float* cU = nullptr;
  if (ci == 0 && cj == 0) { hL = h_ext; cL = c_ext; }
  else if (cj > 0)        { hL = gh_out + (size_t)(k - 1) * BH;
                            cL = gc_out + (size_t)(k - 1) * BH; }
  if (ci > 0)             { hU = gh_out + (size_t)(k - GWN) * BH;
                            cU = gc_out + (size_t)(k - GWN) * BH; }
  const float* hP = grid_h + (size_t)k * BH;
  const float* cP = grid_c + (size_t)k * BH;

  // Stage comb_h / comb_c (16 x 768) as bf16 into LDS.
  for (int idx = tid; idx < 16 * K3; idx += 256) {
    int r = idx / K3, col = idx - r * K3;
    size_t boff = (size_t)(row0 + r) * HN;
    int seg = col >> 8, cc = col & 255;
    float vh, vc;
    if (seg == 0)      { vh = hL ? hL[boff + cc] : 0.0f; vc = cL ? cL[boff + cc] : 0.0f; }
    else if (seg == 1) { vh = hU ? hU[boff + cc] : 0.0f; vc = cU ? cU[boff + cc] : 0.0f; }
    else               { vh = hP[boff + cc];             vc = cP[boff + cc]; }
    sA [idx] = (__bf16)vh;
    sAc[idx] = (__bf16)vc;
  }
  __syncthreads();

  const WT*    Whp_k = W_hp + (size_t)k * HN * K3;
  const WT*    Wcp_k = W_cp + (size_t)k * HN * K3;
  const WT*    Whh_k = W_hh + (size_t)k * G4 * HN;
  const float* bhp_k = b_hp + (size_t)k * HN;
  const float* bcp_k = b_cp + (size_t)k * HN;
  const float* bih_k = b_ih + (size_t)k * G4;
  const float* bhh_k = b_hh + (size_t)k * G4;
  const float* wih_k = W_ih + (size_t)k * G4;

  const __bf16* sArow  = sA  + (size_t)ln * K3;
  const __bf16* sAcrow = sAc + (size_t)ln * K3;

  // -------- Phase 2: ph = comb_h @ W_hp^T + b_hp ; pc = comb_c @ W_cp^T + b_cp
  // Wave w owns h-tiles {2w, 2w+1}. ph -> LDS (bf16) for gates-A; pc stays in regs.
  v8f pcv[2];
#pragma unroll
  for (int t = 0; t < 2; ++t) {
    const int n = (2 * w + t) * 16 + ln;   // output column for this lane
    // ph
    v8f acc = {};
    {
      const WT* wr = Whp_k + (size_t)n * K3 + 16 * kh;
      for (int kk = 0; kk < K3; kk += 32) {
        __builtin_prefetch(wr + kk + 128, 0, 2);
        v16bf a  = ldsA_frag(sArow, kk + 16 * kh);
        v16bf bm = ldB_frag(wr + kk);
        acc = __builtin_amdgcn_wmma_f32_16x16x32_bf16(
                  false, a, false, bm, (short)0, acc, false, false);
      }
    }
    {
      const float bb = bhp_k[n];
#pragma unroll
      for (int v = 0; v < 8; ++v)
        sGA[(v + 8 * kh) * HN + n] = (__bf16)(acc[v] + bb);
    }
    // pc
    v8f accc = {};
    {
      const WT* wr = Wcp_k + (size_t)n * K3 + 16 * kh;
      for (int kk = 0; kk < K3; kk += 32) {
        __builtin_prefetch(wr + kk + 128, 0, 2);
        v16bf a  = ldsA_frag(sAcrow, kk + 16 * kh);
        v16bf bm = ldB_frag(wr + kk);
        accc = __builtin_amdgcn_wmma_f32_16x16x32_bf16(
                   false, a, false, bm, (short)0, accc, false, false);
      }
    }
    {
      const float bb = bcp_k[n];
#pragma unroll
      for (int v = 0; v < 8; ++v) accc[v] += bb;
      pcv[t] = accc;
    }
  }
  __syncthreads();

  // -------- Phase 3: gates = ph @ W_hh^T (+ x*W_ih + b_ih + b_hh), LSTM cell.
  // Wave w computes i/f/g/o gate tiles for its h-tiles {2w, 2w+1} so the
  // elementwise combine is fully in-register (pc is already resident too).
  const __bf16* sGrow = sGA + (size_t)ln * HN;
  float* ghk = gh_out + (size_t)k * BH;
  float* gck = gc_out + (size_t)k * BH;
#pragma unroll
  for (int t = 0; t < 2; ++t) {
    const int ht = 2 * w + t;
    const int nC = ht * 16 + ln;           // h column for this lane
    v8f gi_a = {}, gf_a = {}, gg_a = {}, go_a = {};
    for (int kk = 0; kk < HN; kk += 32) {
      v16bf a = ldsA_frag(sGrow, kk + 16 * kh);
      {
        v16bf bm = ldB_frag(Whh_k + (size_t)(0 * HN + nC) * HN + 16 * kh + kk);
        gi_a = __builtin_amdgcn_wmma_f32_16x16x32_bf16(false, a, false, bm, (short)0, gi_a, false, false);
      }
      {
        v16bf bm = ldB_frag(Whh_k + (size_t)(1 * HN + nC) * HN + 16 * kh + kk);
        gf_a = __builtin_amdgcn_wmma_f32_16x16x32_bf16(false, a, false, bm, (short)0, gf_a, false, false);
      }
      {
        v16bf bm = ldB_frag(Whh_k + (size_t)(2 * HN + nC) * HN + 16 * kh + kk);
        gg_a = __builtin_amdgcn_wmma_f32_16x16x32_bf16(false, a, false, bm, (short)0, gg_a, false, false);
      }
      {
        v16bf bm = ldB_frag(Whh_k + (size_t)(3 * HN + nC) * HN + 16 * kh + kk);
        go_a = __builtin_amdgcn_wmma_f32_16x16x32_bf16(false, a, false, bm, (short)0, go_a, false, false);
      }
    }
    const float wi_i = wih_k[0 * HN + nC], bb_i = bih_k[0 * HN + nC] + bhh_k[0 * HN + nC];
    const float wi_f = wih_k[1 * HN + nC], bb_f = bih_k[1 * HN + nC] + bhh_k[1 * HN + nC];
    const float wi_g = wih_k[2 * HN + nC], bb_g = bih_k[2 * HN + nC] + bhh_k[2 * HN + nC];
    const float wi_o = wih_k[3 * HN + nC], bb_o = bih_k[3 * HN + nC] + bhh_k[3 * HN + nC];
#pragma unroll
    for (int v = 0; v < 8; ++v) {
      const int m = v + 8 * kh;
      const int b = row0 + m;
      const float xv = x[(size_t)b * (GHN * GWN) + k];
      const float gi = gi_a[v] + xv * wi_i + bb_i;
      const float gf = gf_a[v] + xv * wi_f + bb_f;
      const float gg = gg_a[v] + xv * wi_g + bb_g;
      const float go = go_a[v] + xv * wi_o + bb_o;
      const float iS = sigmoidf_(gi);
      const float fS = sigmoidf_(gf);
      const float oS = sigmoidf_(go);
      const float gT = tanhf(gg);
      const float cN = fS * pcv[t][v] + iS * gT;
      const float hN = oS * tanhf(cN);
      const size_t o = (size_t)b * HN + nC;
      ghk[o] = hN;
      gck[o] = cN;
    }
  }
}

// fp32 -> bf16 weight image: 8 floats per iteration, 16B stores.
__global__ __launch_bounds__(256)
void cvt_weights_kernel(const float4* __restrict__ src, v8bf* __restrict__ dst, long n8)
{
  long i = (long)blockIdx.x * blockDim.x + threadIdx.x;
  const long stride = (long)gridDim.x * blockDim.x;
  for (; i < n8; i += stride) {
    float4 a = src[2 * i], b = src[2 * i + 1];
    union { v8bf v; __bf16 e[8]; } u;
    u.e[0] = (__bf16)a.x; u.e[1] = (__bf16)a.y; u.e[2] = (__bf16)a.z; u.e[3] = (__bf16)a.w;
    u.e[4] = (__bf16)b.x; u.e[5] = (__bf16)b.y; u.e[6] = (__bf16)b.z; u.e[7] = (__bf16)b.w;
    dst[i] = u.v;
  }
}

__global__ __launch_bounds__(256)
void lattice_final_kernel(const float* __restrict__ gh, const float* __restrict__ gc,
                          const float* __restrict__ W_out, const float* __restrict__ b_out,
                          float* __restrict__ out, float* __restrict__ fh, float* __restrict__ fc)
{
  const int b = (int)blockIdx.x * (int)blockDim.x + (int)threadIdx.x;
  if (b >= BN) return;
  const size_t last = (size_t)(GHN * GWN - 1) * BN * HN;
  const float* hrow = gh + last + (size_t)b * HN;
  const float* crow = gc + last + (size_t)b * HN;
  float acc = b_out[0];
  for (int n = 0; n < HN; ++n) {
    const float hv = hrow[n];
    acc += hv * W_out[n];
    fh[(size_t)b * HN + n] = hv;
    fc[(size_t)b * HN + n] = crow[n];
  }
  const float r = fmaxf(acc, 0.0f);
  out[b] = 1.0f / (1.0f + __expf(-r));
}

extern "C" void kernel_launch(void* const* d_in, const int* in_sizes, int n_in,
                              void* d_out, int out_size, void* d_ws, size_t ws_size,
                              hipStream_t stream) {
  (void)in_sizes; (void)n_in; (void)out_size;
  const float* x      = (const float*)d_in[0];
  const float* h_ext  = (const float*)d_in[1];
  const float* c_ext  = (const float*)d_in[2];
  const float* grid_h = (const float*)d_in[3];
  const float* grid_c = (const float*)d_in[4];
  const float* W_hp   = (const float*)d_in[5];
  const float* b_hp   = (const float*)d_in[6];
  const float* W_cp   = (const float*)d_in[7];
  const float* b_cp   = (const float*)d_in[8];
  const float* W_ih   = (const float*)d_in[9];
  const float* W_hh   = (const float*)d_in[10];
  const float* b_ih   = (const float*)d_in[11];
  const float* b_hh   = (const float*)d_in[12];
  const float* W_out  = (const float*)d_in[13];
  const float* b_out  = (const float*)d_in[14];

  float* out = (float*)d_out;
  float* fh  = out + 1024;
  float* fc  = out + 263168;
  float* gh  = out + 525312;
  float* gc  = out + 21758976;

  const size_t HPsz = (size_t)GHN * GWN * HN * K3;   // W_hp / W_cp elements
  const size_t HHsz = (size_t)GHN * GWN * G4 * HN;   // W_hh elements
  const size_t need = (2 * HPsz + HHsz) * sizeof(__bf16);
  const bool use_bf16_ws = (ws_size >= need);

  if (use_bf16_ws) {
    // Pre-convert all weights to a bf16 image in d_ws (fits in 192 MB L2).
    __bf16* whp_b = (__bf16*)d_ws;
    __bf16* wcp_b = whp_b + HPsz;
    __bf16* whh_b = wcp_b + HPsz;
    const int cvt_blocks = 2048;
    cvt_weights_kernel<<<cvt_blocks, 256, 0, stream>>>(
        (const float4*)W_hp, (v8bf*)whp_b, (long)(HPsz / 8));
    cvt_weights_kernel<<<cvt_blocks, 256, 0, stream>>>(
        (const float4*)W_cp, (v8bf*)wcp_b, (long)(HPsz / 8));
    cvt_weights_kernel<<<cvt_blocks, 256, 0, stream>>>(
        (const float4*)W_hh, (v8bf*)whh_b, (long)(HHsz / 8));

    for (int d = 0; d < GHN + GWN - 1; ++d) {
      const int ilo = (d > GHN - 1) ? d - (GWN - 1) : 0;
      const int ihi = (d < GHN - 1) ? d : GHN - 1;
      dim3 grid(BN / 16, ihi - ilo + 1);
      lattice_cell_kernel<__bf16><<<grid, 256, 0, stream>>>(
          x, h_ext, c_ext, grid_h, grid_c,
          whp_b, b_hp, wcp_b, b_cp, W_ih, whh_b, b_ih, b_hh,
          gh, gc, d, ilo);
    }
  } else {
    // Fallback: stream fp32 weights, convert in-register in the WMMA loop.
    for (int d = 0; d < GHN + GWN - 1; ++d) {
      const int ilo = (d > GHN - 1) ? d - (GWN - 1) : 0;
      const int ihi = (d < GHN - 1) ? d : GHN - 1;
      dim3 grid(BN / 16, ihi - ilo + 1);
      lattice_cell_kernel<float><<<grid, 256, 0, stream>>>(
          x, h_ext, c_ext, grid_h, grid_c,
          W_hp, b_hp, W_cp, b_cp, W_ih, W_hh, b_ih, b_hh,
          gh, gc, d, ilo);
    }
  }
  lattice_final_kernel<<<(BN + 255) / 256, 256, 0, stream>>>(
      gh, gc, W_out, b_out, out, fh, fc);
}
